// MambaStack_24567212934069
// MI455X (gfx1250) — matrix operand order repaired and checked by reading, not compiled, not run on hardware
//
#include <hip/hip_runtime.h>
#include <hip/hip_bf16.h>

// ---------------- model constants ----------------
#define D_MODEL   512
#define DEPTH     2
#define D_STATE   64
#define D_CONV    4
#define HEADDIM   64
#define D_INNER   1024                    // EXPAND * D_MODEL
#define NHEADS    16                      // D_INNER / HEADDIM
#define CONV_DIM  1152                    // D_INNER + 2*D_STATE
#define D_IN_PROJ 2192                    // 2*D_INNER + 2*D_STATE + NHEADS
#define BATCH     2
#define SEQLEN    1024
#define MROWS     (BATCH * SEQLEN)        // 2048

typedef __attribute__((ext_vector_type(16))) __bf16 v16bf;
typedef __attribute__((ext_vector_type(8)))  float  v8f;

#define WMMA_BF16(a, b, c) \
  __builtin_amdgcn_wmma_f32_16x16x32_bf16(false, (a), false, (b), (short)0, (c), false, false)

// ---------------- elementwise converts ----------------
__global__ void cvt_f32_to_bf16(const float* __restrict__ src,
                                __bf16* __restrict__ dst, int n) {
  int i = blockIdx.x * blockDim.x + threadIdx.x;
  if (i < n) dst[i] = (__bf16)src[i];
}

// W is (K x N) row-major fp32; write Wt as (N x K) bf16 so B-fragments are
// K-contiguous per lane.
__global__ void cvt_w_transpose_bf16(const float* __restrict__ W,
                                     __bf16* __restrict__ Wt, int K, int N) {
  int i = blockIdx.x * blockDim.x + threadIdx.x;
  if (i >= K * N) return;
  int k = i / N, n = i % N;
  Wt[(size_t)n * K + k] = (__bf16)W[(size_t)k * N + n];
}

// ---------------- WMMA bf16 GEMM ----------------
// A: (M x K) bf16 row-major.  Bt: (N x K) bf16 (i.e. B transposed).  C: f32.
// One wave computes a 64x16 tile: 4 M-subtiles share one B fragment ->
// 4 independent WMMAs per K-step (40B/lane/WMMA operand traffic).
// Register double-buffering: K-step k+1 fragments are loaded before the
// WMMAs of step k issue, so loads overlap XDL execution within one wave.
// 8 waves/block -> 512-row strip per block. N tiled by 16 (N=2192=16*137).

// A fragment (16x32, MxK): element i -> K = 16*(lane>>4) + i:
// one contiguous 16-elem (32B) run per lane.
__device__ __forceinline__ v16bf load_a_frag(const __bf16* __restrict__ pr) {
  v16bf a;
#pragma unroll
  for (int i = 0; i < 8; ++i) a[i] = pr[i];
#pragma unroll
  for (int i = 0; i < 8; ++i) a[i + 8] = pr[8 + i];
  return a;
}

// B fragment (32x16, KxN): element i -> K = 16*(lane>>4) + i: one 32B run.
__device__ __forceinline__ v16bf load_b_frag(const __bf16* __restrict__ pr) {
  v16bf b;
#pragma unroll
  for (int i = 0; i < 16; ++i) b[i] = pr[i];
  return b;
}

__global__ void gemm_bf16_wmma(const __bf16* __restrict__ A,
                               const __bf16* __restrict__ Bt,
                               float* __restrict__ C,
                               int M, int N, int K) {
  const int wave = threadIdx.x >> 5;
  const int lane = threadIdx.x & 31;
  const int m0 = blockIdx.y * 512 + wave * 64;   // 64 rows per wave
  const int n0 = blockIdx.x * 16;

  // per-lane base pointers
  const __bf16* pa = A + (size_t)(m0 + (lane & 15)) * K + ((lane >> 4) << 4);
  const __bf16* pb = Bt + (size_t)(n0 + (lane & 15)) * K + ((lane >> 4) << 4);
  const size_t st = (size_t)16 * K;  // 16 rows down

  v8f acc0 = {}, acc1 = {}, acc2 = {}, acc3 = {};

  // prologue: fragments for k0 = 0
  v16bf b0  = load_b_frag(pb);
  v16bf a00 = load_a_frag(pa);
  v16bf a01 = load_a_frag(pa + st);
  v16bf a02 = load_a_frag(pa + 2 * st);
  v16bf a03 = load_a_frag(pa + 3 * st);

  for (int k0 = 0; k0 + 32 < K; k0 += 32) {
    const int kn = k0 + 32;
    // issue next K-step's loads first, then compute on current registers
    v16bf b1  = load_b_frag(pb + kn);
    v16bf a10 = load_a_frag(pa + kn);
    v16bf a11 = load_a_frag(pa + st + kn);
    v16bf a12 = load_a_frag(pa + 2 * st + kn);
    v16bf a13 = load_a_frag(pa + 3 * st + kn);
    // keep-near prefetch ~8 K-steps ahead
    __builtin_prefetch(pa + kn + 224, 0, 3);
    __builtin_prefetch(pb + kn + 224, 0, 3);

    acc0 = WMMA_BF16(a00, b0, acc0);
    acc1 = WMMA_BF16(a01, b0, acc1);
    acc2 = WMMA_BF16(a02, b0, acc2);
    acc3 = WMMA_BF16(a03, b0, acc3);

    b0 = b1; a00 = a10; a01 = a11; a02 = a12; a03 = a13;
  }
  // epilogue: last K-step
  acc0 = WMMA_BF16(a00, b0, acc0);
  acc1 = WMMA_BF16(a01, b0, acc1);
  acc2 = WMMA_BF16(a02, b0, acc2);
  acc3 = WMMA_BF16(a03, b0, acc3);

  // C/D layout: VGPR r -> M = base + r + 8*(lane>>4), N = n0 + (lane&15)
  const int mrow = m0 + ((lane >> 4) << 3);
  const int ncol = n0 + (lane & 15);
#pragma unroll
  for (int r = 0; r < 8; ++r) C[(size_t)(mrow + r) * N + ncol] = acc0[r];
#pragma unroll
  for (int r = 0; r < 8; ++r) C[(size_t)(mrow + 16 + r) * N + ncol] = acc1[r];
#pragma unroll
  for (int r = 0; r < 8; ++r) C[(size_t)(mrow + 32 + r) * N + ncol] = acc2[r];
#pragma unroll
  for (int r = 0; r < 8; ++r) C[(size_t)(mrow + 48 + r) * N + ncol] = acc3[r];
}

// ---------------- dt = softplus(raw + bias) ----------------
__global__ void dt_kernel(const float* __restrict__ zx,
                          const float* __restrict__ dt_bias,
                          float* __restrict__ dt) {
  int i = blockIdx.x * blockDim.x + threadIdx.x;
  if (i >= MROWS * NHEADS) return;
  int row = i >> 4, hh = i & 15;
  float v = zx[(size_t)row * D_IN_PROJ + (D_IN_PROJ - NHEADS) + hh] + dt_bias[hh];
  dt[i] = (v > 20.f) ? v : log1pf(__expf(v));
}

// ---------------- causal conv1d + bias + SiLU ----------------
__global__ void conv_silu_kernel(const float* __restrict__ zx,
                                 const float* __restrict__ w,
                                 const float* __restrict__ bias,
                                 float* __restrict__ out) {
  size_t i = (size_t)blockIdx.x * blockDim.x + threadIdx.x;
  if (i >= (size_t)MROWS * CONV_DIM) return;
  int c = (int)(i % CONV_DIM);
  size_t bl = i / CONV_DIM;
  int l = (int)(bl % SEQLEN);
  int b = (int)(bl / SEQLEN);
  float acc = bias[c];
#pragma unroll
  for (int k = 0; k < D_CONV; ++k) {
    int li = l + k - (D_CONV - 1);
    if (li >= 0)
      acc += zx[((size_t)b * SEQLEN + li) * D_IN_PROJ + D_INNER + c] * w[c * D_CONV + k];
  }
  out[i] = acc / (1.f + __expf(-acc));  // silu
}

// ---------------- sequential SSM scan ----------------
// One block per (batch, head). 256 threads: p = t&63 (head dim),
// ng = t>>6 owns 16 contiguous state columns -> h[64][64] in registers.
__global__ void scan_kernel(const float* __restrict__ xBC,
                            const float* __restrict__ dt,
                            const float* __restrict__ A_log,
                            const float* __restrict__ Dp,
                            float* __restrict__ y) {
  const int b = blockIdx.x >> 4;
  const int h = blockIdx.x & 15;
  const int t = threadIdx.x;
  const int p = t & 63;
  const int ng = t >> 6;

  __shared__ float sx[HEADDIM], sB[D_STATE], sC[D_STATE], sy[4][HEADDIM];
  __shared__ float sdt;

  float hreg[16];
#pragma unroll
  for (int j = 0; j < 16; ++j) hreg[j] = 0.f;

  const float Acoef = -__expf(A_log[h]);
  const float Dh = Dp[h];

  for (int l = 0; l < SEQLEN; ++l) {
    const size_t base = ((size_t)b * SEQLEN + l) * CONV_DIM;
    if (t < 64)        sx[t]        = xBC[base + h * HEADDIM + t];
    else if (t < 128)  sB[t - 64]   = xBC[base + D_INNER + (t - 64)];
    else if (t < 192)  sC[t - 128]  = xBC[base + D_INNER + D_STATE + (t - 128)];
    else if (t == 192) sdt          = dt[((size_t)b * SEQLEN + l) * NHEADS + h];
    __syncthreads();

    const float dtv = sdt;
    const float dA = __expf(dtv * Acoef);
    const float coef = dtv * sx[p];
    float acc = 0.f;
#pragma unroll
    for (int j = 0; j < 16; ++j) {
      const int n = ng * 16 + j;
      hreg[j] = fmaf(dA, hreg[j], coef * sB[n]);
      acc = fmaf(hreg[j], sC[n], acc);
    }
    sy[ng][p] = acc;
    __syncthreads();
    if (t < 64) {
      float yv = sy[0][p] + sy[1][p] + sy[2][p] + sy[3][p] + Dh * sx[p];
      y[((size_t)b * SEQLEN + l) * D_INNER + h * HEADDIM + p] = yv;
    }
    __syncthreads();
  }
}

// ---------------- gate (y * silu(z)) + RMSNorm + bf16 ----------------
__global__ void gate_norm_kernel(const float* __restrict__ y,
                                 const float* __restrict__ zx,
                                 const float* __restrict__ norm_w,
                                 __bf16* __restrict__ out) {
  const int row = blockIdx.x;
  const int t = threadIdx.x;  // 256
  const float* yr = y + (size_t)row * D_INNER;
  const float* zr = zx + (size_t)row * D_IN_PROJ;

  float vals[4];
  float ss = 0.f;
#pragma unroll
  for (int i = 0; i < 4; ++i) {
    int c = t + i * 256;
    float z = zr[c];
    float g = yr[c] * (z / (1.f + __expf(-z)));
    vals[i] = g;
    ss += g * g;
  }
  __shared__ float red[256];
  red[t] = ss;
  __syncthreads();
  for (int s = 128; s > 0; s >>= 1) {
    if (t < s) red[t] += red[t + s];
    __syncthreads();
  }
  const float inv = rsqrtf(red[0] / (float)D_INNER + 1e-5f);
#pragma unroll
  for (int i = 0; i < 4; ++i) {
    int c = t + i * 256;
    out[(size_t)row * D_INNER + c] = (__bf16)(vals[i] * inv * norm_w[c]);
  }
}

// ---------------- host orchestration ----------------
static inline size_t align256(size_t v) { return (v + 255) & ~(size_t)255; }

extern "C" void kernel_launch(void* const* d_in, const int* in_sizes, int n_in,
                              void* d_out, int out_size, void* d_ws, size_t ws_size,
                              hipStream_t stream) {
  (void)in_sizes; (void)n_in; (void)out_size; (void)ws_size;

  const float* x0 = (const float*)d_in[0];
  float* out_final = (float*)d_out;

  // workspace carve-up
  char* ws = (char*)d_ws;
  size_t off = 0;
  float* zx   = (float*)(ws + off); off += align256((size_t)MROWS * D_IN_PROJ * 4);
  float* xBCc = (float*)(ws + off); off += align256((size_t)MROWS * CONV_DIM * 4);
  float* dtb  = (float*)(ws + off); off += align256((size_t)MROWS * NHEADS * 4);
  float* ybuf = (float*)(ws + off); off += align256((size_t)MROWS * D_INNER * 4);
  float* xtmp = (float*)(ws + off); off += align256((size_t)MROWS * D_MODEL * 4);
  __bf16* abf = (__bf16*)(ws + off); off += align256((size_t)MROWS * D_INNER * 2);
  __bf16* wbf = (__bf16*)(ws + off); off += align256((size_t)D_MODEL * D_IN_PROJ * 2);

  const float* cur = x0;
  for (int layer = 0; layer < DEPTH; ++layer) {
    const float* Win    = (const float*)d_in[1 + layer * 8 + 0]; // (512 x 2192)
    const float* conv_w = (const float*)d_in[1 + layer * 8 + 1]; // (1152 x 4)
    const float* conv_b = (const float*)d_in[1 + layer * 8 + 2];
    const float* dtbias = (const float*)d_in[1 + layer * 8 + 3];
    const float* A_log  = (const float*)d_in[1 + layer * 8 + 4];
    const float* Dp     = (const float*)d_in[1 + layer * 8 + 5];
    const float* norm_w = (const float*)d_in[1 + layer * 8 + 6];
    const float* Wout   = (const float*)d_in[1 + layer * 8 + 7]; // (1024 x 512)

    // 1) activations -> bf16
    {
      int n = MROWS * D_MODEL;
      cvt_f32_to_bf16<<<(n + 255) / 256, 256, 0, stream>>>(cur, abf, n);
    }
    // 2) in_proj -> transposed bf16
    {
      int n = D_MODEL * D_IN_PROJ;
      cvt_w_transpose_bf16<<<(n + 255) / 256, 256, 0, stream>>>(Win, wbf, D_MODEL, D_IN_PROJ);
    }
    // 3) zx = x @ Win   (M=2048, N=2192, K=512); 512 rows per block
    {
      dim3 grid(D_IN_PROJ / 16, MROWS / 512);
      gemm_bf16_wmma<<<grid, 256, 0, stream>>>(abf, wbf, zx, MROWS, D_IN_PROJ, D_MODEL);
    }
    // 4) dt
    {
      int n = MROWS * NHEADS;
      dt_kernel<<<(n + 255) / 256, 256, 0, stream>>>(zx, dtbias, dtb);
    }
    // 5) conv + silu
    {
      size_t n = (size_t)MROWS * CONV_DIM;
      conv_silu_kernel<<<(unsigned)((n + 255) / 256), 256, 0, stream>>>(zx, conv_w, conv_b, xBCc);
    }
    // 6) SSM scan (one block per batch*head)
    scan_kernel<<<BATCH * NHEADS, 256, 0, stream>>>(xBCc, dtb, A_log, Dp, ybuf);
    // 7) gate + rmsnorm -> bf16 activations
    gate_norm_kernel<<<MROWS, 256, 0, stream>>>(ybuf, zx, norm_w, abf);
    // 8) out_proj -> transposed bf16
    {
      int n = D_INNER * D_MODEL;
      cvt_w_transpose_bf16<<<(n + 255) / 256, 256, 0, stream>>>(Wout, wbf, D_INNER, D_MODEL);
    }
    // 9) out = y @ Wout  (M=2048, N=512, K=1024)
    {
      float* dst = (layer == DEPTH - 1) ? out_final : xtmp;
      dim3 grid(D_MODEL / 16, MROWS / 512);
      gemm_bf16_wmma<<<grid, 256, 0, stream>>>(abf, wbf, dst, MROWS, D_MODEL, D_INNER);
    }
    cur = xtmp;
  }
}
